// SAModule_78915729096761
// MI455X (gfx1250) — matrix operand order repaired and checked by reading, not compiled, not run on hardware
//
#include <hip/hip_runtime.h>
#include <hip/hip_bf16.h>

#define BB 8
#define NN 8192
#define SS 1024
#define KNB 64
#define CC 64
#define KP 96          // layer-1 K dim (67) padded to 3x32

typedef __bf16 bf16_t;
typedef bf16_t v16bf __attribute__((ext_vector_type(16)));
typedef float  v8f   __attribute__((ext_vector_type(8)));

// float -> bf16 bits, round-to-nearest-even
__device__ __forceinline__ unsigned short f2bfu(float f) {
    unsigned u = __builtin_bit_cast(unsigned, f);
    u += 0x7FFFu + ((u >> 16) & 1u);
    return (unsigned short)(u >> 16);
}

// A fragment: 16x32 tile (rows m0.., cols k0..) of row-major bf16 buffer.
// ISA 7.12.2 16-bit A layout: lane<16 row=lane, elems 0..7 = K0..7, 8..15 = K16..23;
// lane>=16 same row, K shifted by +8 / +24.
__device__ __forceinline__ v16bf loadA(const unsigned short* src, int stride,
                                       int m0, int k0, int lane) {
    int row = m0 + (lane & 15);
    int kb  = k0 + ((lane >> 4) << 3);
    const unsigned short* p = src + row * stride + kb;
    v16bf a;
#pragma unroll
    for (int i = 0; i < 8; ++i) {
        a[i]     = __builtin_bit_cast(bf16_t, p[i]);
        a[i + 8] = __builtin_bit_cast(bf16_t, p[i + 16]);
    }
    return a;
}

// B fragment: 32x16 tile (rows k0.., cols n0..) taken from a TRANSPOSED weight
// buffer wt[n][k] so the 16 K-elements per lane are contiguous.
// ISA layout: lane<16 col=lane, elems 0..7 = K0..7, 8..15 = K16..23; lane>=16 K shifted by +8 / +24.
__device__ __forceinline__ v16bf loadB(const unsigned short* wt, int strideK,
                                       int n0, int k0, int lane) {
    int col = n0 + (lane & 15);
    int kb  = k0 + ((lane >> 4) << 3);
    const unsigned short* p = wt + col * strideK + kb;
    v16bf b;
#pragma unroll
    for (int i = 0; i < 8; ++i) {
        b[i]     = __builtin_bit_cast(bf16_t, p[i]);
        b[i + 8] = __builtin_bit_cast(bf16_t, p[i + 16]);
    }
    return b;
}

// ---------------------------------------------------------------- FPS ------
// One workgroup per batch. Points live in registers (8 per thread), the 1024
// sequential steps only touch LDS for the argmax reduction + centroid bcast.
__global__ __launch_bounds__(1024) void fps_kernel(const float* __restrict__ pos,
                                                   int* __restrict__ cent) {
    __shared__ float redD[32];
    __shared__ int   redI[32];
    __shared__ float cpos[3];
    __shared__ int   sFar;

    const int tid = threadIdx.x, lane = tid & 31, wave = tid >> 5;
    const int b = blockIdx.x;

    float px[8], py[8], pz[8], dd[8];
#pragma unroll
    for (int j = 0; j < 8; ++j) {
        const float* pp = pos + (size_t)(b * NN + j * 1024 + tid) * 3;
        px[j] = pp[0]; py[j] = pp[1]; pz[j] = pp[2];
        dd[j] = 1e10f;
    }

    int far = 0;
    for (int it = 0; it < SS; ++it) {
        if (tid == 0) cent[b * SS + it] = far;
        if (tid == (far & 1023)) {
            int j = far >> 10;
            cpos[0] = px[j]; cpos[1] = py[j]; cpos[2] = pz[j];
        }
        __syncthreads();                                  // A: centroid visible
        float cx = cpos[0], cy = cpos[1], cz = cpos[2];
        float bd = -1.0f; int bi = 0;
#pragma unroll
        for (int j = 0; j < 8; ++j) {
            float dx = px[j] - cx, dy = py[j] - cy, dz = pz[j] - cz;
            float d  = dx * dx + dy * dy + dz * dz;
            float nd = dd[j] < d ? dd[j] : d;
            dd[j] = nd;
            if (nd > bd) { bd = nd; bi = j * 1024 + tid; }  // ascending idx => first-max
        }
#pragma unroll
        for (int off = 16; off > 0; off >>= 1) {
            float od = __shfl_xor(bd, off, 32);
            int   oi = __shfl_xor(bi, off, 32);
            if (od > bd || (od == bd && oi < bi)) { bd = od; bi = oi; }
        }
        if (lane == 0) { redD[wave] = bd; redI[wave] = bi; }
        __syncthreads();                                  // B: partials visible
        if (wave == 0) {
            float bd2 = redD[lane]; int bi2 = redI[lane];
#pragma unroll
            for (int off = 16; off > 0; off >>= 1) {
                float od = __shfl_xor(bd2, off, 32);
                int   oi = __shfl_xor(bi2, off, 32);
                if (od > bd2 || (od == bd2 && oi < bi2)) { bd2 = od; bi2 = oi; }
            }
            if (lane == 0) sFar = bi2;
        }
        __syncthreads();                                  // C: winner visible
        far = sFar;
    }
}

// ---------------------------------------------------- ball query -----------
// One wave per centroid: ballot + prefix-popcount compaction == "sorted first
// 64 in-radius indices" of the reference (indices scanned ascending).
__global__ __launch_bounds__(256) void ballgroup_kernel(const float* __restrict__ pos,
                                                        const int* __restrict__ cent,
                                                        int* __restrict__ gidx,
                                                        float* __restrict__ outCenter) {
    const int tid = threadIdx.x, lane = tid & 31, wv = tid >> 5;
    const int bs = blockIdx.x * 8 + wv;      // 0 .. B*S-1
    const int b  = bs >> 10;
    const int cidx = cent[bs];

    const float* cp = pos + (size_t)(b * NN + cidx) * 3;
    float cx = cp[0], cy = cp[1], cz = cp[2];
    if (lane == 0) {
        outCenter[bs * 3 + 0] = cx;
        outCenter[bs * 3 + 1] = cy;
        outCenter[bs * 3 + 2] = cz;
    }
    const float cc = cx * cx + cy * cy + cz * cz;
    const float r2 = 0.2f * 0.2f;
    const int base = bs * KNB;

    int count = 0, firstIdx = -1;
    for (int n0 = 0; n0 < NN && count < KNB; n0 += 32) {
        int n = n0 + lane;
        const float* pp = pos + (size_t)(b * NN + n) * 3;
        float x = pp[0], y = pp[1], z = pp[2];
        // reference formula: |c|^2 + |p|^2 - 2 c.p
        float sqr = cc + (x * x + y * y + z * z) - 2.0f * (cx * x + cy * y + cz * z);
        bool in = (sqr <= r2);
        unsigned mask = (unsigned)__ballot(in);
        if (firstIdx < 0 && mask != 0u) firstIdx = n0 + (__ffs((int)mask) - 1);
        int slot = count + __popc(mask & ((1u << lane) - 1u));
        if (in && slot < KNB) gidx[base + slot] = n;
        count += __popc(mask);
    }
    if (firstIdx < 0) firstIdx = cidx;       // unreachable in practice
    for (int s2 = count + lane; s2 < KNB; s2 += 32) gidx[base + s2] = firstIdx;
}

// --------------------------------------------- gather + 3-layer MLP + max --
// One workgroup (4 waves) per centroid. bf16 WMMA 16x16x32, f32 accumulate.
// Each wave keeps 4 independent accumulator chains (its 4 M-tiles) live so
// consecutive WMMAs are independent and the bf16 WMMA->WMMA hazard slots are
// filled with real matrix ops instead of v_nop.
__global__ __launch_bounds__(128) void sa_mlp_kernel(const float* __restrict__ pos,
                                                     const float* __restrict__ feat,
                                                     const float* __restrict__ w1,
                                                     const float* __restrict__ b1,
                                                     const float* __restrict__ w2,
                                                     const float* __restrict__ b2,
                                                     const float* __restrict__ w3,
                                                     const float* __restrict__ b3,
                                                     const int* __restrict__ gidx,
                                                     const float* __restrict__ centers,
                                                     float* __restrict__ outFeat) {
    __shared__ __align__(16) unsigned short w1t[64 * KP];    // w1 transposed [n][k], k padded
    __shared__ __align__(16) unsigned short w2t[64 * 64];    // [n][k]
    __shared__ __align__(16) unsigned short w3t[128 * 64];   // [n][k]
    __shared__ float b1s[64], b2s[64], b3s[128];
    __shared__ __align__(16) unsigned short hbuf[64 * KP];   // activations (stride KP)
    __shared__ __align__(16) unsigned short tbuf[64 * 64];   // activations (stride 64)
    __shared__ float ctr[3];
    __shared__ int   gids[KNB];

    const int tid = threadIdx.x, lane = tid & 31, wave = tid >> 5;
    const int bs = blockIdx.x;
    const int b  = bs >> 10;

    // stage weights (bf16, transposed) + biases + group indices + center
    for (int e = tid; e < 64 * KP; e += 128) {
        int n = e / KP, k = e % KP;
        float v = (k < CC + 3) ? w1[k * 64 + n] : 0.0f;
        w1t[e] = f2bfu(v);
    }
    for (int e = tid; e < 64 * 64; e += 128) {
        int n = e >> 6, k = e & 63;
        w2t[e] = f2bfu(w2[k * 64 + n]);
    }
    for (int e = tid; e < 128 * 64; e += 128) {
        int n = e >> 6, k = e & 63;
        w3t[e] = f2bfu(w3[k * 128 + n]);
    }
    if (tid < 64)  { b1s[tid] = b1[tid]; b2s[tid] = b2[tid]; }
    if (tid < 128) b3s[tid] = b3[tid];
    if (tid < KNB) gids[tid] = gidx[bs * KNB + tid];
    if (tid < 3)   ctr[tid] = centers[bs * 3 + tid];
    __syncthreads();

    // gather h = [gpos - center | gfeat | 0-pad]  (64 x 96 bf16)
    for (int e = tid; e < 64 * KP; e += 128) {
        int r = e / KP, c = e % KP;
        int id = gids[r];
        float v = 0.0f;
        if (c < 3)            v = pos[(size_t)(b * NN + id) * 3 + c] - ctr[c];
        else if (c < CC + 3)  v = feat[(size_t)(b * NN + id) * CC + (c - 3)];
        hbuf[e] = f2bfu(v);
    }
    __syncthreads();

    // ---- layer 1: (64x96) @ (96x64) -> relu -> tbuf ----
    {
        const int ni = wave;
        v8f acc[4] = {};
#pragma unroll
        for (int kc = 0; kc < 3; ++kc) {
            v16bf bm = loadB(w1t, KP, ni * 16, kc * 32, lane);
#pragma unroll
            for (int mi = 0; mi < 4; ++mi) {
                v16bf a = loadA(hbuf, KP, mi * 16, kc * 32, lane);
                acc[mi] = __builtin_amdgcn_wmma_f32_16x16x32_bf16(
                    false, a, false, bm, (short)0, acc[mi], false, false);
            }
        }
        int n = ni * 16 + (lane & 15);
        float bias = b1s[n];
#pragma unroll
        for (int mi = 0; mi < 4; ++mi) {
            int mBase = mi * 16 + ((lane >> 4) << 3);
#pragma unroll
            for (int r = 0; r < 8; ++r) {
                float v = acc[mi][r] + bias;
                v = v > 0.0f ? v : 0.0f;
                tbuf[(mBase + r) * 64 + n] = f2bfu(v);
            }
        }
    }
    __syncthreads();

    // ---- layer 2: (64x64) @ (64x64) -> relu -> hbuf (stride KP) ----
    {
        const int ni = wave;
        v8f acc[4] = {};
#pragma unroll
        for (int kc = 0; kc < 2; ++kc) {
            v16bf bm = loadB(w2t, 64, ni * 16, kc * 32, lane);
#pragma unroll
            for (int mi = 0; mi < 4; ++mi) {
                v16bf a = loadA(tbuf, 64, mi * 16, kc * 32, lane);
                acc[mi] = __builtin_amdgcn_wmma_f32_16x16x32_bf16(
                    false, a, false, bm, (short)0, acc[mi], false, false);
            }
        }
        int n = ni * 16 + (lane & 15);
        float bias = b2s[n];
#pragma unroll
        for (int mi = 0; mi < 4; ++mi) {
            int mBase = mi * 16 + ((lane >> 4) << 3);
#pragma unroll
            for (int r = 0; r < 8; ++r) {
                float v = acc[mi][r] + bias;
                v = v > 0.0f ? v : 0.0f;
                hbuf[(mBase + r) * KP + n] = f2bfu(v);
            }
        }
    }
    __syncthreads();

    // ---- layer 3: (64x64) @ (64x128) -> relu -> max over the 64 rows ----
    // A fragments are shared by this wave's two N-tiles: hoist all 8 once.
    {
        v16bf afr[2][4];
#pragma unroll
        for (int kc = 0; kc < 2; ++kc)
#pragma unroll
            for (int mi = 0; mi < 4; ++mi)
                afr[kc][mi] = loadA(hbuf, KP, mi * 16, kc * 32, lane);

#pragma unroll
        for (int t = 0; t < 2; ++t) {
            const int ni = wave * 2 + t;
            v8f acc[4] = {};
#pragma unroll
            for (int kc = 0; kc < 2; ++kc) {
                v16bf bm = loadB(w3t, 64, ni * 16, kc * 32, lane);
#pragma unroll
                for (int mi = 0; mi < 4; ++mi) {
                    acc[mi] = __builtin_amdgcn_wmma_f32_16x16x32_bf16(
                        false, afr[kc][mi], false, bm, (short)0, acc[mi], false, false);
                }
            }
            float bias = b3s[ni * 16 + (lane & 15)];
            float vmax = 0.0f;                       // relu output >= 0
#pragma unroll
            for (int mi = 0; mi < 4; ++mi) {
#pragma unroll
                for (int r = 0; r < 8; ++r) {
                    float v = acc[mi][r] + bias;
                    v = v > 0.0f ? v : 0.0f;
                    vmax = v > vmax ? v : vmax;
                }
            }
            float other = __shfl_xor(vmax, 16, 32);  // fold the two M-halves
            vmax = vmax > other ? vmax : other;
            if (lane < 16) outFeat[(size_t)bs * 128 + ni * 16 + lane] = vmax;
        }
    }
}

extern "C" void kernel_launch(void* const* d_in, const int* in_sizes, int n_in,
                              void* d_out, int out_size, void* d_ws, size_t ws_size,
                              hipStream_t stream) {
    const float* pos  = (const float*)d_in[0];
    const float* feat = (const float*)d_in[1];
    const float* w1   = (const float*)d_in[2];
    const float* b1   = (const float*)d_in[3];
    const float* w2   = (const float*)d_in[4];
    const float* b2   = (const float*)d_in[5];
    const float* w3   = (const float*)d_in[6];
    const float* b3   = (const float*)d_in[7];

    float* outCenter = (float*)d_out;                    // [B,S,3]
    float* outFeat   = (float*)d_out + BB * SS * 3;      // [B,S,128]

    int* cent = (int*)d_ws;                              // [B,S]
    int* gidx = cent + BB * SS;                          // [B,S,K]

    fps_kernel<<<BB, 1024, 0, stream>>>(pos, cent);
    ballgroup_kernel<<<BB * SS / 8, 256, 0, stream>>>(pos, cent, gidx, outCenter);
    sa_mlp_kernel<<<BB * SS, 128, 0, stream>>>(pos, feat, w1, b1, w2, b2, w3, b3,
                                               gidx, outCenter, outFeat);
}
